// FinancialTransformer_18794776887824
// MI455X (gfx1250) — matrix operand order repaired and checked by reading, not compiled, not run on hardware
//
#include <hip/hip_runtime.h>
#include <hip/hip_bf16.h>

typedef _Float16 half_t;
typedef __attribute__((ext_vector_type(16))) _Float16 v16h;
typedef __attribute__((ext_vector_type(8)))  _Float16 v8h;
typedef __attribute__((ext_vector_type(4)))  _Float16 v4h;
typedef __attribute__((ext_vector_type(2)))  _Float16 v2h;
typedef __attribute__((ext_vector_type(8)))  float    v8f;

union U16h { v16h v; v8h h[2]; };

#define WMMA_F16(a, b, c) \
  __builtin_amdgcn_wmma_f32_16x16x32_f16(false, (a), false, (b), (short)0, (c), false, false)

// Problem constants
#define BB   8
#define SS   1024
#define FIN  32
#define DD   512
#define HH   8
#define LL   6
#define DFF  2048
#define DKK  64
#define NCLS 3
#define MM   (BB * SS)   // 8192 token rows

// ---------------------------------------------------------------------------
// Tiled GEMM: Y[M,N] = act( X[M,K] @ W[K,N] + bias[N] )  (+ pe[row%S,:] if pe)
// f32 in/out, f16 WMMA compute, f32 accumulate.
// Block = 256 threads (8 waves). Tile 128x128, BK=32. Wave tile 32x64.
// Register-double-buffered staging: next K-tile's global loads are issued
// before the current tile's WMMAs, hiding HBM latency behind the matrix pipe.
// ---------------------------------------------------------------------------
#define BM 128
#define BN 128
#define BK 32
#define LDA 40   // halves per A row   (32 + 8 pad) -> 80 B pitch, 16B-aligned
#define LDBT 40  // halves per B^T row (32 + 8 pad) -> 80 B pitch, 16B-aligned

__device__ __forceinline__ v16h load_fragA(const half_t* base, int koff)
{
    // halves {koff..koff+15} contiguous -> v16h (A-operand chunking)
    U16h u;
    u.h[0] = *(const v8h*)(base + koff);
    u.h[1] = *(const v8h*)(base + koff + 8);
    return u.v;
}

__device__ __forceinline__ v16h load_fragB(const half_t* base, int koff)
{
    // halves {koff..koff+15} contiguous -> v16h (B-operand from transposed LDS)
    U16h u;
    u.h[0] = *(const v8h*)(base + koff);
    u.h[1] = *(const v8h*)(base + koff + 8);
    return u.v;
}

__global__ __launch_bounds__(256)
void gemm_wmma(const float* __restrict__ X, const float* __restrict__ W,
               const float* __restrict__ bias, const float* __restrict__ pe,
               float* __restrict__ Y, int M, int K, int N, int act)
{
    __shared__ __align__(16) half_t As[BM * LDA];    // 10 KB
    __shared__ __align__(16) half_t BsT[BN * LDBT];  // 10 KB  (B transposed: [n][k])

    const int tid  = threadIdx.x;
    const int lane = tid & 31;
    const int w    = tid >> 5;
    const int wr   = w >> 1;      // 0..3
    const int wc   = w & 1;       // 0..1
    const int bm   = blockIdx.y * BM;
    const int bn   = blockIdx.x * BN;

    const int l15     = lane & 15;
    const int halfSel = lane >> 4;
    const int akoff   = halfSel * 16;
    const int bkoff   = halfSel * 16;

    // Per-thread staging coordinates (fixed across K steps)
    const int ar0 = tid >> 2;           // A row (segment 0); segment 1 = ar0+64
    const int ac  = (tid & 3) * 8;      // A col start (8 consecutive floats)
    const int nb  = tid & 127;          // B column
    const int kq  = (tid >> 7) * 4;     // B k-quad base; it-th quad at kq + it*8

    const float* xp0 = X + (size_t)(bm + ar0) * K + ac;
    const float* xp1 = xp0 + (size_t)64 * K;
    const float* wpb = W + bn + nb;

    v8f acc[2][4];
#pragma unroll
    for (int m = 0; m < 2; ++m)
#pragma unroll
        for (int n = 0; n < 4; ++n) acc[m][n] = (v8f){};

    // Staging registers (double buffer lives here, not in LDS)
    float4 av[4];
    float  wv[16];

    // ---- load K-tile k0 into registers: all loads issued before any use ----
    auto load_tiles = [&](int k0) {
        av[0] = *(const float4*)(xp0 + k0);
        av[1] = *(const float4*)(xp0 + k0 + 4);
        av[2] = *(const float4*)(xp1 + k0);
        av[3] = *(const float4*)(xp1 + k0 + 4);
#pragma unroll
        for (int it = 0; it < 4; ++it)
#pragma unroll
            for (int j = 0; j < 4; ++j)
                wv[it * 4 + j] = wpb[(size_t)(k0 + kq + it * 8 + j) * N];
    };
    // ---- convert staged registers to f16 and store to LDS ----
    auto store_tiles = [&]() {
        v8h p0, p1;
        p0[0] = (half_t)av[0].x; p0[1] = (half_t)av[0].y; p0[2] = (half_t)av[0].z; p0[3] = (half_t)av[0].w;
        p0[4] = (half_t)av[1].x; p0[5] = (half_t)av[1].y; p0[6] = (half_t)av[1].z; p0[7] = (half_t)av[1].w;
        p1[0] = (half_t)av[2].x; p1[1] = (half_t)av[2].y; p1[2] = (half_t)av[2].z; p1[3] = (half_t)av[2].w;
        p1[4] = (half_t)av[3].x; p1[5] = (half_t)av[3].y; p1[6] = (half_t)av[3].z; p1[7] = (half_t)av[3].w;
        *(v8h*)&As[ar0 * LDA + ac]        = p0;
        *(v8h*)&As[(ar0 + 64) * LDA + ac] = p1;
#pragma unroll
        for (int it = 0; it < 4; ++it) {
            v4h qd;
            qd[0] = (half_t)wv[it * 4 + 0];
            qd[1] = (half_t)wv[it * 4 + 1];
            qd[2] = (half_t)wv[it * 4 + 2];
            qd[3] = (half_t)wv[it * 4 + 3];
            *(v4h*)&BsT[nb * LDBT + kq + it * 8] = qd;
        }
    };

    load_tiles(0);
    for (int k0 = 0; k0 < K; k0 += BK) {
        store_tiles();
        __syncthreads();
        if (k0 + BK < K) load_tiles(k0 + BK);   // overlaps with WMMAs below

        v16h afr[2];
#pragma unroll
        for (int mt = 0; mt < 2; ++mt) {
            int row = wr * 32 + mt * 16 + l15;
            afr[mt] = load_fragA(&As[row * LDA], akoff);
        }
#pragma unroll
        for (int nt = 0; nt < 4; ++nt) {
            int col = wc * 64 + nt * 16 + l15;
            v16h bfr = load_fragB(&BsT[col * LDBT], bkoff);
#pragma unroll
            for (int mt = 0; mt < 2; ++mt)
                acc[mt][nt] = WMMA_F16(afr[mt], bfr, acc[mt][nt]);
        }
        __syncthreads();
    }

    // Epilogue: bias (+pe) (+relu)
#pragma unroll
    for (int mt = 0; mt < 2; ++mt) {
#pragma unroll
        for (int nt = 0; nt < 4; ++nt) {
            int col = bn + wc * 64 + nt * 16 + l15;
            float bv = bias ? bias[col] : 0.f;
#pragma unroll
            for (int r = 0; r < 8; ++r) {
                int row = bm + wr * 32 + mt * 16 + halfSel * 8 + r;
                float val = acc[mt][nt][r] + bv;
                if (pe) val += pe[(size_t)(row & (SS - 1)) * DD + col]; // N==DD here
                if (act) val = val > 0.f ? val : 0.f;
                Y[(size_t)row * N + col] = val;
            }
        }
    }
}

// ---------------------------------------------------------------------------
// V transpose: vT[b,h,d,s] = v[b,s,h*64+d]   (makes P@V B-fragments contiguous)
// ---------------------------------------------------------------------------
__global__ __launch_bounds__(256)
void transpose_v(const float* __restrict__ v, float* __restrict__ vT)
{
    __shared__ float tile[64][65];
    const int bid = blockIdx.x;
    const int s0  = (bid & 15) * 64;     // S/64 = 16
    const int h   = (bid >> 4) & 7;
    const int b   = bid >> 7;
    const int tid = threadIdx.x;

#pragma unroll
    for (int e = tid; e < 64 * 64; e += 256) {
        int r = e >> 6, c = e & 63;      // r = s, c = d
        tile[r][c] = v[(size_t)(b * SS + s0 + r) * DD + h * DKK + c];
    }
    __syncthreads();
#pragma unroll
    for (int e = tid; e < 64 * 64; e += 256) {
        int r = e >> 6, c = e & 63;      // r = d, c = s
        vT[((size_t)((b * HH + h) * DKK + r)) * SS + s0 + c] = tile[c][r];
    }
}

// ---------------------------------------------------------------------------
// Attention: per block = one (b, head, 32-query tile).
// Full 32x1024 score strip in LDS (f32). WMMA for QK^T and P*V.
// Block = 256 threads (8 waves). Dynamic LDS ~133 KB (WGP has 320 KB).
// ---------------------------------------------------------------------------
#define QB 32
#define SC_LD 1028   // f32 stride for score rows (4-float (16 B) multiple, bank-skewed)
#define QT_LD 72     // f16 stride for staged Q tile (144 B pitch, 16B multiple)

__global__ __launch_bounds__(256)
void attention_wmma(const float* __restrict__ q, const float* __restrict__ k,
                    const float* __restrict__ vT, float* __restrict__ out)
{
    extern __shared__ __align__(16) char smem[];
    float*  sc = (float*)smem;                                         // [QB][SC_LD]
    half_t* qt = (half_t*)(smem + (size_t)QB * SC_LD * sizeof(float)); // [QB][QT_LD]

    const int tid  = threadIdx.x;
    const int lane = tid & 31;
    const int w    = tid >> 5;
    const int bid  = blockIdx.x;
    const int qblk = bid & 31;           // S/QB = 32 tiles
    const int h    = (bid >> 5) & 7;
    const int b    = bid >> 8;
    const int q0   = qblk * QB;
    const int h0   = h * DKK;

    const int l15     = lane & 15;
    const int halfSel = lane >> 4;
    const int akoff   = halfSel * 16;
    const int bkoff   = halfSel * 16;

    // Stage Q tile (QB x DK) as f16
#pragma unroll
    for (int e = tid * 2; e < QB * DKK; e += 512) {
        int r = e >> 6, c = e & 63;
        const float* qp = &q[(size_t)(b * SS + q0 + r) * DD + h0 + c];
        v2h p; p[0] = (half_t)qp[0]; p[1] = (half_t)qp[1];
        *(v2h*)&qt[r * QT_LD + c] = p;
    }
    __syncthreads();

    // ---- Phase 1: scores = (Q K^T)/8 ; each wave owns a 128-key chunk ----
    const int key_base = w * 128;
#pragma unroll 2
    for (int kt = 0; kt < 8; ++kt) {
        int key0 = key_base + kt * 16;
        v8f acc0 = (v8f){}, acc1 = (v8f){};
#pragma unroll
        for (int ks = 0; ks < 2; ++ks) {
            // B fragment: 16 consecutive d-values of one key row (64B aligned)
            const float4* kp =
                (const float4*)(k + (size_t)(b * SS + key0 + l15) * DD + h0 + ks * 32 + bkoff);
            float4 f0 = kp[0], f1 = kp[1], f2 = kp[2], f3 = kp[3];
            v16h bfr;
            bfr[0]  = (half_t)f0.x; bfr[1]  = (half_t)f0.y; bfr[2]  = (half_t)f0.z; bfr[3]  = (half_t)f0.w;
            bfr[4]  = (half_t)f1.x; bfr[5]  = (half_t)f1.y; bfr[6]  = (half_t)f1.z; bfr[7]  = (half_t)f1.w;
            bfr[8]  = (half_t)f2.x; bfr[9]  = (half_t)f2.y; bfr[10] = (half_t)f2.z; bfr[11] = (half_t)f2.w;
            bfr[12] = (half_t)f3.x; bfr[13] = (half_t)f3.y; bfr[14] = (half_t)f3.z; bfr[15] = (half_t)f3.w;

            v16h af0 = load_fragA(&qt[(0 * 16 + l15) * QT_LD + ks * 32], akoff);
            v16h af1 = load_fragA(&qt[(1 * 16 + l15) * QT_LD + ks * 32], akoff);
            acc0 = WMMA_F16(af0, bfr, acc0);
            acc1 = WMMA_F16(af1, bfr, acc1);
        }
#pragma unroll
        for (int r = 0; r < 8; ++r) {
            sc[(0 * 16 + halfSel * 8 + r) * SC_LD + key0 + l15] = acc0[r] * 0.125f;
            sc[(1 * 16 + halfSel * 8 + r) * SC_LD + key0 + l15] = acc1[r] * 0.125f;
        }
    }
    __syncthreads();

    // ---- Phase 2: row softmax (f32, wave32 shuffles) ----
#pragma unroll 1
    for (int rr = 0; rr < 4; ++rr) {
        int row = w * 4 + rr;
        float mx = -3.4e38f;
        for (int j = lane; j < SS; j += 32) mx = fmaxf(mx, sc[row * SC_LD + j]);
#pragma unroll
        for (int off = 16; off; off >>= 1) mx = fmaxf(mx, __shfl_xor(mx, off));
        float sum = 0.f;
        for (int j = lane; j < SS; j += 32) {
            float e = __expf(sc[row * SC_LD + j] - mx);
            sc[row * SC_LD + j] = e;
            sum += e;
        }
#pragma unroll
        for (int off = 16; off; off >>= 1) sum += __shfl_xor(sum, off);
        float inv = 1.f / sum;
        for (int j = lane; j < SS; j += 32) sc[row * SC_LD + j] *= inv;
    }
    __syncthreads();

    // ---- Phase 3: ctx = P @ V ; 8 waves = 2 mtiles x 4 ntiles ----
    {
        const int mt = w >> 2;   // 0..1
        const int nt = w & 3;    // 0..3
        const int col = nt * 16 + l15;
        const float* vrow = vT + ((size_t)((b * HH + h) * DKK + col)) * SS;
        const float* prow = sc + (size_t)(mt * 16 + l15) * SC_LD;
        v8f accc = (v8f){};
#pragma unroll 2
        for (int ks = 0; ks < SS / 32; ++ks) {
            // A fragment: P row chunk {akoff..akoff+15} (f32 LDS -> f16)
            const float* ap = prow + ks * 32;
            float4 a0 = *(const float4*)(ap + akoff);
            float4 a1 = *(const float4*)(ap + akoff + 4);
            float4 a2 = *(const float4*)(ap + akoff + 8);
            float4 a3 = *(const float4*)(ap + akoff + 12);
            v16h af;
            af[0]  = (half_t)a0.x; af[1]  = (half_t)a0.y; af[2]  = (half_t)a0.z; af[3]  = (half_t)a0.w;
            af[4]  = (half_t)a1.x; af[5]  = (half_t)a1.y; af[6]  = (half_t)a1.z; af[7]  = (half_t)a1.w;
            af[8]  = (half_t)a2.x; af[9]  = (half_t)a2.y; af[10] = (half_t)a2.z; af[11] = (half_t)a2.w;
            af[12] = (half_t)a3.x; af[13] = (half_t)a3.y; af[14] = (half_t)a3.z; af[15] = (half_t)a3.w;

            // B fragment: 16 consecutive keys of one V^T row (64B aligned)
            const float4* vp = (const float4*)(vrow + ks * 32 + bkoff);
            float4 f0 = vp[0], f1 = vp[1], f2 = vp[2], f3 = vp[3];
            v16h bf;
            bf[0]  = (half_t)f0.x; bf[1]  = (half_t)f0.y; bf[2]  = (half_t)f0.z; bf[3]  = (half_t)f0.w;
            bf[4]  = (half_t)f1.x; bf[5]  = (half_t)f1.y; bf[6]  = (half_t)f1.z; bf[7]  = (half_t)f1.w;
            bf[8]  = (half_t)f2.x; bf[9]  = (half_t)f2.y; bf[10] = (half_t)f2.z; bf[11] = (half_t)f2.w;
            bf[12] = (half_t)f3.x; bf[13] = (half_t)f3.y; bf[14] = (half_t)f3.z; bf[15] = (half_t)f3.w;

            accc = WMMA_F16(af, bf, accc);
        }
#pragma unroll
        for (int r = 0; r < 8; ++r) {
            int row = q0 + mt * 16 + halfSel * 8 + r;
            out[(size_t)(b * SS + row) * DD + h0 + nt * 16 + l15] = accc[r];
        }
    }
}

// ---------------------------------------------------------------------------
// h = LayerNorm(h + src) * g + b  (one row of D=512 per block, 256 threads)
// ---------------------------------------------------------------------------
__global__ __launch_bounds__(256)
void add_layernorm(float* __restrict__ hbuf, const float* __restrict__ src,
                   const float* __restrict__ g, const float* __restrict__ bta)
{
    __shared__ float red[16];
    const int row = blockIdx.x;
    const int tid = threadIdx.x;
    const size_t base = (size_t)row * DD;

    float x0 = hbuf[base + tid]       + src[base + tid];
    float x1 = hbuf[base + tid + 256] + src[base + tid + 256];
    float s  = x0 + x1;
    float sq = x0 * x0 + x1 * x1;
#pragma unroll
    for (int off = 16; off; off >>= 1) {
        s  += __shfl_xor(s,  off);
        sq += __shfl_xor(sq, off);
    }
    if ((tid & 31) == 0) { red[tid >> 5] = s; red[8 + (tid >> 5)] = sq; }
    __syncthreads();
    float S = 0.f, SQ = 0.f;
#pragma unroll
    for (int i = 0; i < 8; ++i) { S += red[i]; SQ += red[8 + i]; }
    float mean = S * (1.f / DD);
    float var  = SQ * (1.f / DD) - mean * mean;
    float inv  = rsqrtf(var + 1e-5f);
    hbuf[base + tid]       = (x0 - mean) * inv * g[tid]       + bta[tid];
    hbuf[base + tid + 256] = (x1 - mean) * inv * g[tid + 256] + bta[tid + 256];
}

// ---------------------------------------------------------------------------
// pooled[b,d] = mean_s h[b,s,d]
// ---------------------------------------------------------------------------
__global__ void pool_mean(const float* __restrict__ hbuf, float* __restrict__ pooled)
{
    int b = blockIdx.x >> 1;
    int d = (blockIdx.x & 1) * 256 + threadIdx.x;
    float s = 0.f;
    for (int sIdx = 0; sIdx < SS; ++sIdx)
        s += hbuf[(size_t)(b * SS + sIdx) * DD + d];
    pooled[b * DD + d] = s * (1.f / SS);
}

// ---------------------------------------------------------------------------
// Tiny FC head: out[b,j] = act(in[b,:] @ W[:,j] + bias[j])
// ---------------------------------------------------------------------------
__global__ void fc_kernel(const float* __restrict__ in, const float* __restrict__ wgt,
                          const float* __restrict__ bias, float* __restrict__ out,
                          int K, int N, int act)
{
    int b = blockIdx.x;
    int j = threadIdx.x;
    if (j < N) {
        float s = bias[j];
        for (int i = 0; i < K; ++i) s += in[b * K + i] * wgt[i * N + j];
        out[b * N + j] = act ? fmaxf(s, 0.f) : s;
    }
}

// ---------------------------------------------------------------------------
// Host orchestration
// ---------------------------------------------------------------------------
extern "C" void kernel_launch(void* const* d_in, const int* in_sizes, int n_in,
                              void* d_out, int out_size, void* d_ws, size_t ws_size,
                              hipStream_t stream)
{
    const float* x     = (const float*)d_in[0];
    const float* pe    = (const float*)d_in[1];
    const float* emb_w = (const float*)d_in[2];
    const float* emb_b = (const float*)d_in[3];
    const float* Wq    = (const float*)d_in[4];
    const float* bq    = (const float*)d_in[5];
    const float* Wk    = (const float*)d_in[6];
    const float* bk    = (const float*)d_in[7];
    const float* Wv    = (const float*)d_in[8];
    const float* bv    = (const float*)d_in[9];
    const float* Wo    = (const float*)d_in[10];
    const float* bo    = (const float*)d_in[11];
    const float* ln1_g = (const float*)d_in[12];
    const float* ln1_b = (const float*)d_in[13];
    const float* ln2_g = (const float*)d_in[14];
    const float* ln2_b = (const float*)d_in[15];
    const float* W1    = (const float*)d_in[16];
    const float* b1    = (const float*)d_in[17];
    const float* W2    = (const float*)d_in[18];
    const float* b2    = (const float*)d_in[19];
    const float* fc1_w = (const float*)d_in[20];
    const float* fc1_b = (const float*)d_in[21];
    const float* fc2_w = (const float*)d_in[22];
    const float* fc2_b = (const float*)d_in[23];
    const float* fc3_w = (const float*)d_in[24];
    const float* fc3_b = (const float*)d_in[25];

    float* ws = (float*)d_ws;
    const size_t MD = (size_t)MM * DD;     // 8192*512
    float* hbuf  = ws;
    float* buf1  = hbuf + MD;              // q / attn_out / start of DFF region
    float* buf2  = buf1 + MD;              // k
    float* buf3  = buf2 + MD;              // v
    float* buf4  = buf3 + MD;              // ctx
    float* buf5  = buf4 + MD;              // vT during attention / ffn output
    float* ffact = buf1;                   // 8192 x 2048 spans buf1..buf4
    float* pooled = buf5 + MD;
    float* z1     = pooled + BB * DD;
    float* z2     = z1 + BB * 256;

    const dim3 blk(256);
    const dim3 gD(DD / BN, MM / BM);       // N=512 tiles
    const dim3 gF1(DFF / BN, MM / BM);     // N=2048 tiles

    const int attn_smem = QB * SC_LD * (int)sizeof(float) + QB * QT_LD * (int)sizeof(half_t);
    (void)hipFuncSetAttribute((const void*)attention_wmma,
                              hipFuncAttributeMaxDynamicSharedMemorySize, attn_smem);

    // Embedding + positional encoding: h = x @ emb_w + emb_b + pe
    gemm_wmma<<<gD, blk, 0, stream>>>(x, emb_w, emb_b, pe, hbuf, MM, FIN, DD, 0);

    for (int l = 0; l < LL; ++l) {
        const size_t wo = (size_t)l * DD * DD;
        // Q, K, V projections
        gemm_wmma<<<gD, blk, 0, stream>>>(hbuf, Wq + wo, bq + l * DD, nullptr, buf1, MM, DD, DD, 0);
        gemm_wmma<<<gD, blk, 0, stream>>>(hbuf, Wk + wo, bk + l * DD, nullptr, buf2, MM, DD, DD, 0);
        gemm_wmma<<<gD, blk, 0, stream>>>(hbuf, Wv + wo, bv + l * DD, nullptr, buf3, MM, DD, DD, 0);
        // V transpose for contiguous P@V fragments
        transpose_v<<<dim3(BB * HH * (SS / 64)), blk, 0, stream>>>(buf3, buf5);
        // Attention: ctx -> buf4
        attention_wmma<<<dim3(BB * HH * (SS / QB)), blk, attn_smem, stream>>>(buf1, buf2, buf5, buf4);
        // Output projection: attn_out -> buf1
        gemm_wmma<<<gD, blk, 0, stream>>>(buf4, Wo + wo, bo + l * DD, nullptr, buf1, MM, DD, DD, 0);
        // h = LN(h + attn_out)
        add_layernorm<<<dim3(MM), blk, 0, stream>>>(hbuf, buf1, ln1_g + l * DD, ln1_b + l * DD);
        // FFN
        gemm_wmma<<<gF1, blk, 0, stream>>>(hbuf, W1 + (size_t)l * DD * DFF, b1 + l * DFF,
                                           nullptr, ffact, MM, DD, DFF, 1);
        gemm_wmma<<<gD, blk, 0, stream>>>(ffact, W2 + (size_t)l * DFF * DD, b2 + l * DD,
                                          nullptr, buf5, MM, DFF, DD, 0);
        // h = LN(h + ffn)
        add_layernorm<<<dim3(MM), blk, 0, stream>>>(hbuf, buf5, ln2_g + l * DD, ln2_b + l * DD);
    }

    // Head
    pool_mean<<<dim3(BB * 2), blk, 0, stream>>>(hbuf, pooled);
    fc_kernel<<<dim3(BB), dim3(256), 0, stream>>>(pooled, fc1_w, fc1_b, z1, DD, 256, 1);
    fc_kernel<<<dim3(BB), dim3(128), 0, stream>>>(z1, fc2_w, fc2_b, z2, 256, 128, 1);
    fc_kernel<<<dim3(BB), dim3(32), 0, stream>>>(z2, fc3_w, fc3_b, (float*)d_out, 128, NCLS, 0);
}
